// StrategyAttention_55748675502545
// MI455X (gfx1250) — matrix operand order repaired and checked by reading, not compiled, not run on hardware
//
#include <hip/hip_runtime.h>
#include <hip/hip_bf16.h>

#define B_  64
#define S_  2048
#define H_  512
#define H2_ 1024

typedef __attribute__((ext_vector_type(16))) __bf16 v16bf;
typedef __attribute__((ext_vector_type(8)))  __bf16 v8bf;
typedef __attribute__((ext_vector_type(8)))  float  v8f;

// ---------------------------------------------------------------------------
// Convert W[:, 0:H] (f32, row stride 2H) -> bf16 Wk[n][k], n-major, k contiguous.
// Row-contiguous [n][k] directly gives the B fragments this code uses (A matched) for
// v_wmma_f32_16x16x32_bf16 (lane = column n, 16 contiguous K per half-wave).
// ---------------------------------------------------------------------------
__global__ void convert_wk(const float* __restrict__ W, __bf16* __restrict__ Wk) {
    int idx = blockIdx.x * blockDim.x + threadIdx.x;   // one float4 per thread
    int n = (idx << 2) >> 9;                           // / 512
    int k = (idx << 2) & 511;
    float4 w = *(const float4*)(W + (size_t)n * H2_ + k);
    __bf16* dst = Wk + (size_t)n * H_ + k;
    dst[0] = (__bf16)w.x; dst[1] = (__bf16)w.y;
    dst[2] = (__bf16)w.z; dst[3] = (__bf16)w.w;
}

// ---------------------------------------------------------------------------
// bias[b,d] = sum_h W[d, H+h] * q[b,h]   (tiny: 33 MFLOP total)
// ---------------------------------------------------------------------------
__global__ void bias_kernel(const float* __restrict__ W, const float* __restrict__ q,
                            float* __restrict__ bias) {
    int idx = blockIdx.x * blockDim.x + threadIdx.x;   // [0, B*H)
    int b = idx >> 9;
    int d = idx & 511;
    const float4* wr = (const float4*)(W + (size_t)d * H2_ + H_);
    const float4* qb = (const float4*)(q + (size_t)b * H_);
    float s = 0.f;
    #pragma unroll 4
    for (int i = 0; i < H_ / 4; ++i) {
        float4 w = wr[i], x = qb[i];
        s += w.x * x.x + w.y * x.y + w.z * x.z + w.w * x.w;
    }
    bias[idx] = s;
}

// ---------------------------------------------------------------------------
// score[b,s] = sum_d v[d] * tanh( (keys[b,s,:] . Wk[d,:]) + bias[b,d] )
// WG = 256 threads (8 waves): 32 s-rows x 512 d-cols, K=512 in 16 steps of 32.
// B fragments use a copy-free ping-pong double buffer (2 K-steps per loop
// iteration), so global-load latency overlaps 8 WMMAs with no v_mov rotation
// and no WMMA->VALU hazard NOPs.
// ---------------------------------------------------------------------------
__launch_bounds__(256, 1)
__global__ void score_kernel(const float* __restrict__ keys,
                             const __bf16* __restrict__ Wk,
                             const float* __restrict__ v,
                             const float* __restrict__ bias,
                             float* __restrict__ out) {
    constexpr int MROWS = 32;
    constexpr int LDA   = H_ + 8;                      // pad to spread LDS banks
    __shared__ __align__(16) __bf16 aTile[MROWS][LDA]; // 33.3 KB
    __shared__ float sScratch[8][MROWS];

    const int tid  = threadIdx.x;
    const int lane = tid & 31;
    const int wave = tid >> 5;
    const int hh   = lane >> 4;        // half-wave: 0 or 1
    const int l16  = lane & 15;

    const int tile = blockIdx.x;       // B * (S/32) tiles
    const int b    = tile >> 6;        // 64 tiles per batch
    const int srow = (tile & 63) * MROWS;

    // Stage keys tile (32 x 512 f32) -> bf16 in LDS (read keys exactly once).
    {
        const float4* src = (const float4*)(keys + ((size_t)b * S_ + srow) * H_);
        for (int i = tid; i < MROWS * (H_ / 4); i += 256) {
            int r = i >> 7;                    // / 128
            int c4 = i & 127;
            float4 x = src[(size_t)r * (H_ / 4) + c4];
            __bf16* d = &aTile[r][c4 << 2];
            d[0] = (__bf16)x.x; d[1] = (__bf16)x.y;
            d[2] = (__bf16)x.z; d[3] = (__bf16)x.w;
        }
    }
    __syncthreads();

    const int npanel = wave * 64;
    const __bf16* wkBase = Wk + (size_t)(npanel + l16) * H_ + hh * 16;

    v8f acc[2][4];
    #pragma unroll
    for (int mt = 0; mt < 2; ++mt)
        #pragma unroll
        for (int nt = 0; nt < 4; ++nt)
            #pragma unroll
            for (int j = 0; j < 8; ++j)
                acc[mt][nt][j] = 0.f;

    // B-fragment loader: 4 N-tiles at K-step ksv.
    auto loadB = [&](int ksv, v16bf* dst) {
        #pragma unroll
        for (int nt = 0; nt < 4; ++nt)
            dst[nt] = *(const v16bf*)(wkBase + (size_t)(nt * 16) * H_ + ksv);
    };
    // One K-step: A fragments from LDS (row = lane%16; contiguous K order to
    // match B: lanes 0-15 hold K 0-15, lanes 16-31 hold K 16-31), then
    // 8 WMMAs against the given B buffer.
    auto step = [&](int ksv, const v16bf* bfr) {
        const int ka = ksv + hh * 16;
        v16bf afr[2];
        #pragma unroll
        for (int mt = 0; mt < 2; ++mt) {
            const int row = mt * 16 + l16;
            v8bf lo = *(const v8bf*)&aTile[row][ka];
            v8bf hi = *(const v8bf*)&aTile[row][ka + 8];
            v16bf a;
            #pragma unroll
            for (int j = 0; j < 8; ++j) { a[j] = lo[j]; a[j + 8] = hi[j]; }
            afr[mt] = a;
        }
        #pragma unroll
        for (int nt = 0; nt < 4; ++nt)
            #pragma unroll
            for (int mt = 0; mt < 2; ++mt)
                acc[mt][nt] = __builtin_amdgcn_wmma_f32_16x16x32_bf16(
                    false, afr[mt], false, bfr[nt], (short)0, acc[mt][nt],
                    false, false);
    };

    v16bf b0[4], b1[4];
    loadB(0, b0);                                   // prologue
    for (int ks = 0; ks < H_; ks += 64) {           // 2 K-steps per iteration
        loadB(ks + 32, b1);                         // prefetch odd step
        step(ks, b0);                               // compute even step
        loadB((ks + 64) & (H_ - 1), b0);            // prefetch next even step
        step(ks + 32, b1);                          // compute odd step
    }

    // Fused epilogue: + bias, tanh, * v[d], reduce over d.
    // C layout: lane L holds N = L%16; VGPR r holds M = r (lanes 0-15) or
    // M = r+8 (lanes 16-31).
    const float* biasb = bias + (size_t)b * H_;
    float vn[4], bn[4];
    #pragma unroll
    for (int nt = 0; nt < 4; ++nt) {
        const int n = npanel + nt * 16 + l16;
        vn[nt] = v[n];
        bn[nt] = biasb[n];
    }
    #pragma unroll
    for (int mt = 0; mt < 2; ++mt) {
        float part[8];
        #pragma unroll
        for (int r = 0; r < 8; ++r) {
            float p = 0.f;
            #pragma unroll
            for (int nt = 0; nt < 4; ++nt)
                p += tanhf(acc[mt][nt][r] + bn[nt]) * vn[nt];
            // reduce the 16 lanes of each half-wave (same row, different N)
            #pragma unroll
            for (int off = 1; off < 16; off <<= 1)
                p += __shfl_xor(p, off, 32);
            part[r] = p;
        }
        if (l16 == 0) {
            #pragma unroll
            for (int r = 0; r < 8; ++r)
                sScratch[wave][mt * 16 + hh * 8 + r] = part[r];
        }
    }
    __syncthreads();
    if (tid < MROWS) {
        float s = 0.f;
        #pragma unroll
        for (int w = 0; w < 8; ++w) s += sScratch[w][tid];
        out[(size_t)b * S_ + srow + tid] = s;
    }
}

// ---------------------------------------------------------------------------
// Per batch: softmax over S, then new_q[b,h] = sum_s attn[s] * keys[b,s,h].
// Weighted sum parallelized 8x over S-chunks, float4 h-stripes per lane.
// ---------------------------------------------------------------------------
__global__ void softmax_newq_kernel(const float* __restrict__ scores,
                                    const float* __restrict__ keys,
                                    float* __restrict__ newq) {
    const int b = blockIdx.x;
    const int tid = threadIdx.x;
    __shared__ float attn[S_];
    __shared__ float red[256];
    __shared__ float part[8][H_];

    float loc[8];
    float m = -1e30f;
    #pragma unroll
    for (int i = 0; i < 8; ++i) {
        loc[i] = scores[(size_t)b * S_ + tid + i * 256];
        m = fmaxf(m, loc[i]);
    }
    red[tid] = m; __syncthreads();
    for (int off = 128; off > 0; off >>= 1) {
        if (tid < off) red[tid] = fmaxf(red[tid], red[tid + off]);
        __syncthreads();
    }
    m = red[0]; __syncthreads();

    float s = 0.f;
    #pragma unroll
    for (int i = 0; i < 8; ++i) { loc[i] = __expf(loc[i] - m); s += loc[i]; }
    red[tid] = s; __syncthreads();
    for (int off = 128; off > 0; off >>= 1) {
        if (tid < off) red[tid] += red[tid + off];
        __syncthreads();
    }
    const float inv = 1.f / red[0];
    #pragma unroll
    for (int i = 0; i < 8; ++i) attn[tid + i * 256] = loc[i] * inv;
    __syncthreads();

    // Weighted sum: chunk = 256 s-rows, lane owns a 16-wide h-stripe.
    const int chunk = tid >> 5;          // 0..7
    const int laneh = tid & 31;          // 0..31
    const int hbase = laneh * 16;
    float a[16];
    #pragma unroll
    for (int j = 0; j < 16; ++j) a[j] = 0.f;

    const float* kb = keys + ((size_t)b * S_ + (size_t)chunk * 256) * H_ + hbase;
    const float* at = &attn[chunk * 256];
    for (int sI = 0; sI < 256; ++sI) {
        const float w = at[sI];
        const float4* row = (const float4*)(kb + (size_t)sI * H_);
        #pragma unroll
        for (int j = 0; j < 4; ++j) {
            float4 x = row[j];
            a[j * 4 + 0] += w * x.x;
            a[j * 4 + 1] += w * x.y;
            a[j * 4 + 2] += w * x.z;
            a[j * 4 + 3] += w * x.w;
        }
    }
    #pragma unroll
    for (int j = 0; j < 16; ++j) part[chunk][hbase + j] = a[j];
    __syncthreads();

    // Reduce the 8 chunk-partials; each thread finalizes h = tid, tid+256.
    float s0 = 0.f, s1 = 0.f;
    #pragma unroll
    for (int c = 0; c < 8; ++c) {
        s0 += part[c][tid];
        s1 += part[c][tid + 256];
    }
    newq[(size_t)b * H_ + tid]       = s0;
    newq[(size_t)b * H_ + tid + 256] = s1;
}

// ---------------------------------------------------------------------------
extern "C" void kernel_launch(void* const* d_in, const int* in_sizes, int n_in,
                              void* d_out, int out_size, void* d_ws, size_t ws_size,
                              hipStream_t stream) {
    (void)in_sizes; (void)n_in; (void)out_size; (void)ws_size;
    const float* keys  = (const float*)d_in[0];
    const float* query = (const float*)d_in[1];
    const float* v1    = (const float*)d_in[2];
    const float* W1    = (const float*)d_in[3];
    const float* v2    = (const float*)d_in[4];
    const float* W2    = (const float*)d_in[5];
    float* out = (float*)d_out;

    char* ws = (char*)d_ws;                       // ~1.9 MB used
    __bf16* W1k    = (__bf16*)(ws);               // 512 KB
    __bf16* W2k    = (__bf16*)(ws + 524288);      // 512 KB
    float*  bias1  = (float*)(ws + 1048576);      // 128 KB
    float*  bias2  = (float*)(ws + 1179648);      // 128 KB
    float*  scores = (float*)(ws + 1310720);      // 512 KB
    float*  newq   = (float*)(ws + 1835008);      // 128 KB

    convert_wk<<<256, 256, 0, stream>>>(W1, W1k);
    convert_wk<<<256, 256, 0, stream>>>(W2, W2k);
    bias_kernel<<<128, 256, 0, stream>>>(W1, query, bias1);
    score_kernel<<<B_ * (S_ / 32), 256, 0, stream>>>(keys, W1k, v1, bias1, scores);
    softmax_newq_kernel<<<B_, 256, 0, stream>>>(scores, keys, newq);
    bias_kernel<<<128, 256, 0, stream>>>(W2, newq, bias2);
    score_kernel<<<B_ * (S_ / 32), 256, 0, stream>>>(keys, W2k, v2, bias2, out);
}
